// SuccessFieldModule_22471268892849
// MI455X (gfx1250) — matrix operand order repaired and checked by reading, not compiled, not run on hardware
//
#include <hip/hip_runtime.h>
#include <math.h>

typedef __attribute__((ext_vector_type(16))) _Float16 v16h;
typedef __attribute__((ext_vector_type(2)))  _Float16 h2v;
typedef __attribute__((ext_vector_type(8)))  float    v8f;

#define B_    256
#define FD_   1024
#define HD_   256
#define H3_   128
#define GG_   4096      // G*G
#define MT_   128       // M-rows per workgroup tile (8 waves * 16 rows)
#define NTILES_ ((B_ * GG_) / MT_)   // 8192
#define NWG_  1024

// ---------------------------------------------------------------------------
// Context encoder: features[256,1024] -> z_c[256,64]  (tiny, plain VALU)
// ---------------------------------------------------------------------------
__global__ void __launch_bounds__(256)
encoder_kernel(const float* __restrict__ feat,
               const float* __restrict__ w1, const float* __restrict__ b1,
               const float* __restrict__ w2, const float* __restrict__ b2,
               const float* __restrict__ w3, const float* __restrict__ b3,
               float* __restrict__ zc)
{
    __shared__ float sf[FD_];
    __shared__ float sh1[HD_];
    __shared__ float sh2[HD_ / 2];
    const int b = blockIdx.x, t = threadIdx.x;
    for (int i = t; i < FD_; i += 256) sf[i] = feat[b * FD_ + i];
    __syncthreads();
    {
        float acc = b1[t];
        for (int k = 0; k < FD_; ++k) acc = fmaf(sf[k], w1[k * HD_ + t], acc);
        sh1[t] = fmaxf(acc, 0.f);
    }
    __syncthreads();
    if (t < 128) {
        float acc = b2[t];
        for (int k = 0; k < HD_; ++k) acc = fmaf(sh1[k], w2[k * 128 + t], acc);
        sh2[t] = fmaxf(acc, 0.f);
    }
    __syncthreads();
    if (t < 64) {
        float acc = b3[t];
        for (int k = 0; k < 128; ++k) acc = fmaf(sh2[k], w3[k * 64 + t], acc);
        zc[b * 64 + t] = acc;
    }
}

// ---------------------------------------------------------------------------
// zpart[b,n] = f16( sum_k z_c[b,k]*sf_w1[2+k,n] + sf_b1[n] )   [256,256] f16
// ---------------------------------------------------------------------------
__global__ void __launch_bounds__(256)
zpart_kernel(const float* __restrict__ zc, const float* __restrict__ sfw1,
             const float* __restrict__ sfb1, _Float16* __restrict__ zpart)
{
    __shared__ float sz[64];
    const int b = blockIdx.x, n = threadIdx.x;
    if (n < 64) sz[n] = zc[b * 64 + n];
    __syncthreads();
    float acc = sfb1[n];
    for (int k = 0; k < 64; ++k) acc = fmaf(sz[k], sfw1[(2 + k) * HD_ + n], acc);
    zpart[b * HD_ + n] = (_Float16)acc;
}

// ---------------------------------------------------------------------------
// cpart[p,n] = f16( coords[p,0]*sf_w1[0,n] + coords[p,1]*sf_w1[1,n] )
// ---------------------------------------------------------------------------
__global__ void __launch_bounds__(256)
cpart_kernel(const float* __restrict__ coords, const float* __restrict__ sfw1,
             _Float16* __restrict__ cpart)
{
    const int i = blockIdx.x * 256 + threadIdx.x;
    const int p = i >> 8, n = i & 255;
    cpart[i] = (_Float16)(coords[p * 2] * sfw1[n] + coords[p * 2 + 1] * sfw1[HD_ + n]);
}

// ---------------------------------------------------------------------------
// Pack row-major f32 W[K][N] into CDNA5 WMMA f16 B-fragment tiles.
// Tile (nt, kb) is 32x16 (KxN), 512 halves, kb contiguous within nt.
// In-tile half index = lane*16 + j*2 + h; lane<16 holds K=0..15 of column
// (nt*16 + lane&15), lane>=16 holds K=16..31.  V_j = K{2j, 2j+1}.
// ---------------------------------------------------------------------------
__global__ void __launch_bounds__(256)
pack_b_kernel(const float* __restrict__ W, _Float16* __restrict__ P, int K, int N)
{
    const int i = blockIdx.x * 256 + threadIdx.x;
    if (i >= K * N) return;
    const int tile = i >> 9;
    const int rem  = i & 511;
    const int l = rem >> 4;
    const int j = (rem >> 1) & 7;
    const int h = rem & 1;
    const int kbN = K >> 5;
    const int kb = tile % kbN;
    const int nt = tile / kbN;
    const int k = (kb << 5) + ((l >> 4) << 4) + (j << 1) + h;
    const int n = (nt << 4) + (l & 15);
    P[i] = (_Float16)W[k * N + n];
}

// Pack w4[128] replicated into all 16 columns of 4 B-tiles (K blocks of 32).
__global__ void __launch_bounds__(256)
pack_w4_kernel(const float* __restrict__ w4, _Float16* __restrict__ P)
{
    const int i = blockIdx.x * 256 + threadIdx.x;   // 0..2047
    if (i >= 2048) return;
    const int kb = i >> 9;
    const int rem = i & 511;
    const int l = rem >> 4;
    const int j = (rem >> 1) & 7;
    const int h = rem & 1;
    const int k = (kb << 5) + ((l >> 4) << 4) + (j << 1) + h;
    P[i] = (_Float16)w4[k];
}

// ---------------------------------------------------------------------------
// Main fused field kernel
// ---------------------------------------------------------------------------
union AFrag { v16h v; unsigned int u[8]; };

#define H2_STRIDE 264   // 256 + 8 halves pad (bank de-conflict)
#define H3_STRIDE 136   // 128 + 8 halves pad

// dynamic LDS layout (bytes) — staging buffers FIRST so ds offsets fit imm16
#define SM_H2   0
#define SM_H3   (SM_H2 + 8 * 16 * H2_STRIDE * 2)   // +67584
#define SM_B2   (SM_H3 + 8 * 16 * H3_STRIDE * 2)   // +34816
#define SM_B3   (SM_B2 + 1024)
#define SM_B4   (SM_B3 + 512)
#define SM_W4T  (SM_B4 + 64)
#define SM_W2   (SM_W4T + 4096)
#define SM_W3   (SM_W2 + 131072)
#define SM_TOTAL (SM_W3 + 65536)                    // 304,704 bytes < 320 KB

#define WMMA16(a, b, c) __builtin_amdgcn_wmma_f32_16x16x32_f16( \
    false, (a), false, (b), (short)0, (c), false, false)

// CDNA5 async DMA: global -> LDS, 16 bytes per lane, tracked by ASYNCcnt.
// GVS addressing: lds_addr VGPR, 32-bit per-lane offset VGPR, 64-bit SGPR base.
__device__ __forceinline__ void async_g2l_b128(unsigned lds_addr, unsigned goff,
                                               unsigned long long gbase)
{
    asm volatile("global_load_async_to_lds_b128 %0, %1, %2"
                 :: "v"(lds_addr), "v"(goff), "s"(gbase) : "memory");
}

__global__ void __launch_bounds__(256)
field_kernel(const _Float16* __restrict__ cpart, // [4096][256] f16
             const _Float16* __restrict__ zpart, // [256][256]  f16
             const _Float16* __restrict__ w2p,   // packed B tiles, 65536 halves
             const _Float16* __restrict__ w3p,   // packed B tiles, 32768 halves
             const _Float16* __restrict__ w4p,   // packed B tiles, 2048 halves
             const float* __restrict__ b2,       // [256]
             const float* __restrict__ b3,       // [128]
             const float* __restrict__ b4,       // [1]
             float* __restrict__ out)            // [256*4096]
{
    extern __shared__ char smem[];
    _Float16* s_h2  = (_Float16*)(smem + SM_H2);
    _Float16* s_h3  = (_Float16*)(smem + SM_H3);
    float*    s_b2  = (float*)(smem + SM_B2);
    float*    s_b3  = (float*)(smem + SM_B3);
    float*    s_b4  = (float*)(smem + SM_B4);
    _Float16* s_w4t = (_Float16*)(smem + SM_W4T);
    _Float16* s_w2  = (_Float16*)(smem + SM_W2);
    _Float16* s_w3  = (_Float16*)(smem + SM_W3);

    const int tid  = threadIdx.x;
    const int lane = tid & 31;
    const int wv   = tid >> 5;
    const int lhi  = lane >> 4;   // 0: lanes 0-15, 1: lanes 16-31
    const int lr   = lane & 15;

    // ---- stage packed weights into LDS with async global->LDS DMA ----
    // (no static __shared__, so dynamic LDS starts at byte 0 of the
    //  wave's allocation: SM_* constants are the hardware LDS addresses)
    {
        for (int i = tid; i < 131072 / 16; i += 256)            // W2: 8192 ops
            async_g2l_b128(SM_W2 + i * 16, i * 16, (unsigned long long)w2p);
        for (int i = tid; i < 65536 / 16; i += 256)             // W3: 4096 ops
            async_g2l_b128(SM_W3 + i * 16, i * 16, (unsigned long long)w3p);
        if (tid < 4096 / 16)                                    // W4 tiles
            async_g2l_b128(SM_W4T + tid * 16, tid * 16, (unsigned long long)w4p);
        // biases via normal path while DMAs are in flight
        s_b2[tid] = b2[tid];
        if (tid < 128) s_b3[tid] = b3[tid];
        if (tid == 0) s_b4[0] = b4[0];
        asm volatile("s_wait_asynccnt 0x0" ::: "memory");
    }
    __syncthreads();

    _Float16* h2w = s_h2 + wv * (16 * H2_STRIDE);
    _Float16* h3w = s_h3 + wv * (16 * H3_STRIDE);
    const float bias4 = s_b4[0];
    const h2v  hzero = {(_Float16)0.0f, (_Float16)0.0f};

    for (int t = blockIdx.x; t < NTILES_; t += gridDim.x) {
        const int b    = t >> 5;            // 32 tiles per batch row
        const int p0   = (t & 31) * MT_;
        const int row0 = p0 + wv * 16;      // this wave's first grid point

        // prefetch next tile's A-inputs into caches while we crunch this one
        {
            const int tn = t + gridDim.x;
            if (tn < NTILES_) {
                const int rn = (tn & 31) * MT_ + wv * 16 + lr;
                __builtin_prefetch(&cpart[(size_t)rn * HD_ + (lhi << 7)], 0, 1);
            }
        }

        // ---- build layer-1 output (h1) directly as WMMA A fragments ----
        // h1 = relu(cpart[p] + zpart[b])  (packed f16 math)
        AFrag A[8];
        const _Float16* crow = cpart + (size_t)(row0 + lr) * HD_;
        const _Float16* zrow = zpart + b * HD_;
#pragma unroll
        for (int kb = 0; kb < 8; ++kb) {
#pragma unroll
            for (int j = 0; j < 8; ++j) {
                const int k = (kb << 5) + (lhi << 4) + (j << 1);
                h2v c = *(const h2v*)&crow[k];
                h2v z = *(const h2v*)&zrow[k];
                h2v r = __builtin_elementwise_max(c + z, hzero);
                A[kb].u[j] = __builtin_bit_cast(unsigned int, r);
            }
        }

        // ---- layer 2: h2[16,256] = relu(h1 @ W2 + b2) ----
        // 4 concurrent N-tile accumulator chains, B frags double-buffered
        for (int nt = 0; nt < 16; nt += 4) {
            v8f acc0 = {0.f,0.f,0.f,0.f,0.f,0.f,0.f,0.f};
            v8f acc1 = acc0, acc2 = acc0, acc3 = acc0;
            const _Float16* bt = s_w2 + (nt * 8) * 512 + lane * 16;
            v16h bf0 = *(const v16h*)(bt);
            v16h bf1 = *(const v16h*)(bt + 4096);
            v16h bf2 = *(const v16h*)(bt + 8192);
            v16h bf3 = *(const v16h*)(bt + 12288);
#pragma unroll
            for (int kb = 0; kb < 8; ++kb) {
                v16h nf0, nf1, nf2, nf3;
                if (kb < 7) {
                    nf0 = *(const v16h*)(bt + (kb + 1) * 512);
                    nf1 = *(const v16h*)(bt + (kb + 1) * 512 + 4096);
                    nf2 = *(const v16h*)(bt + (kb + 1) * 512 + 8192);
                    nf3 = *(const v16h*)(bt + (kb + 1) * 512 + 12288);
                } else {
                    nf0 = bf0; nf1 = bf1; nf2 = bf2; nf3 = bf3;
                }
                acc0 = WMMA16(A[kb].v, bf0, acc0);
                acc1 = WMMA16(A[kb].v, bf1, acc1);
                acc2 = WMMA16(A[kb].v, bf2, acc2);
                acc3 = WMMA16(A[kb].v, bf3, acc3);
                bf0 = nf0; bf1 = nf1; bf2 = nf2; bf3 = nf3;
            }
            const float bi0 = s_b2[nt * 16 + lr];
            const float bi1 = s_b2[nt * 16 + 16 + lr];
            const float bi2 = s_b2[nt * 16 + 32 + lr];
            const float bi3 = s_b2[nt * 16 + 48 + lr];
            _Float16* hp = h2w + (lhi * 8) * H2_STRIDE + nt * 16 + lr;
#pragma unroll
            for (int v = 0; v < 8; ++v) {
                hp[v * H2_STRIDE]      = (_Float16)fmaxf(acc0[v] + bi0, 0.f);
                hp[v * H2_STRIDE + 16] = (_Float16)fmaxf(acc1[v] + bi1, 0.f);
                hp[v * H2_STRIDE + 32] = (_Float16)fmaxf(acc2[v] + bi2, 0.f);
                hp[v * H2_STRIDE + 48] = (_Float16)fmaxf(acc3[v] + bi3, 0.f);
            }
        }

        // ---- re-fragment h2 from LDS into A layout ----
#pragma unroll
        for (int kb = 0; kb < 8; ++kb) {
#pragma unroll
            for (int j = 0; j < 8; ++j) {
                const int k = (kb << 5) + (lhi << 4) + (j << 1);
                A[kb].u[j] = *(const unsigned int*)&h2w[lr * H2_STRIDE + k];
            }
        }

        // ---- layer 3: h3[16,128] = relu(h2 @ W3 + b3) ----
        for (int nt = 0; nt < 8; nt += 4) {
            v8f acc0 = {0.f,0.f,0.f,0.f,0.f,0.f,0.f,0.f};
            v8f acc1 = acc0, acc2 = acc0, acc3 = acc0;
            const _Float16* bt = s_w3 + (nt * 8) * 512 + lane * 16;
            v16h bf0 = *(const v16h*)(bt);
            v16h bf1 = *(const v16h*)(bt + 4096);
            v16h bf2 = *(const v16h*)(bt + 8192);
            v16h bf3 = *(const v16h*)(bt + 12288);
#pragma unroll
            for (int kb = 0; kb < 8; ++kb) {
                v16h nf0, nf1, nf2, nf3;
                if (kb < 7) {
                    nf0 = *(const v16h*)(bt + (kb + 1) * 512);
                    nf1 = *(const v16h*)(bt + (kb + 1) * 512 + 4096);
                    nf2 = *(const v16h*)(bt + (kb + 1) * 512 + 8192);
                    nf3 = *(const v16h*)(bt + (kb + 1) * 512 + 12288);
                } else {
                    nf0 = bf0; nf1 = bf1; nf2 = bf2; nf3 = bf3;
                }
                acc0 = WMMA16(A[kb].v, bf0, acc0);
                acc1 = WMMA16(A[kb].v, bf1, acc1);
                acc2 = WMMA16(A[kb].v, bf2, acc2);
                acc3 = WMMA16(A[kb].v, bf3, acc3);
                bf0 = nf0; bf1 = nf1; bf2 = nf2; bf3 = nf3;
            }
            const float bi0 = s_b3[nt * 16 + lr];
            const float bi1 = s_b3[nt * 16 + 16 + lr];
            const float bi2 = s_b3[nt * 16 + 32 + lr];
            const float bi3 = s_b3[nt * 16 + 48 + lr];
            _Float16* hp = h3w + (lhi * 8) * H3_STRIDE + nt * 16 + lr;
#pragma unroll
            for (int v = 0; v < 8; ++v) {
                hp[v * H3_STRIDE]      = (_Float16)fmaxf(acc0[v] + bi0, 0.f);
                hp[v * H3_STRIDE + 16] = (_Float16)fmaxf(acc1[v] + bi1, 0.f);
                hp[v * H3_STRIDE + 32] = (_Float16)fmaxf(acc2[v] + bi2, 0.f);
                hp[v * H3_STRIDE + 48] = (_Float16)fmaxf(acc3[v] + bi3, 0.f);
            }
        }

        // ---- layer 4 via WMMA: logits[16] = h3 @ w4 (w4 replicated in all
        //      16 B columns, so every lane of a half-wave holds all 8 rows) ----
#pragma unroll
        for (int kb = 0; kb < 4; ++kb) {
#pragma unroll
            for (int j = 0; j < 8; ++j) {
                const int k = (kb << 5) + (lhi << 4) + (j << 1);
                A[kb].u[j] = *(const unsigned int*)&h3w[lr * H3_STRIDE + k];
            }
        }
        {
            v8f acc = {0.f,0.f,0.f,0.f,0.f,0.f,0.f,0.f};
            const _Float16* bt = s_w4t + lane * 16;
            v16h bf = *(const v16h*)(bt);
#pragma unroll
            for (int kb = 0; kb < 4; ++kb) {
                v16h nf = (kb < 3) ? *(const v16h*)(bt + (kb + 1) * 512) : bf;
                acc = WMMA16(A[kb].v, bf, acc);
                bf = nf;
            }
            if (lr == 0) {   // lanes 0 (rows 0-7) and 16 (rows 8-15) store
                float4 o0, o1;
                o0.x = 1.f / (1.f + __expf(-(acc[0] + bias4)));
                o0.y = 1.f / (1.f + __expf(-(acc[1] + bias4)));
                o0.z = 1.f / (1.f + __expf(-(acc[2] + bias4)));
                o0.w = 1.f / (1.f + __expf(-(acc[3] + bias4)));
                o1.x = 1.f / (1.f + __expf(-(acc[4] + bias4)));
                o1.y = 1.f / (1.f + __expf(-(acc[5] + bias4)));
                o1.z = 1.f / (1.f + __expf(-(acc[6] + bias4)));
                o1.w = 1.f / (1.f + __expf(-(acc[7] + bias4)));
                float* op = out + (size_t)b * GG_ + row0 + lhi * 8;
                *(float4*)(op)     = o0;
                *(float4*)(op + 4) = o1;
            }
        }
    }
}

// ---------------------------------------------------------------------------
extern "C" void kernel_launch(void* const* d_in, const int* in_sizes, int n_in,
                              void* d_out, int out_size, void* d_ws, size_t ws_size,
                              hipStream_t stream)
{
    (void)in_sizes; (void)n_in; (void)out_size; (void)ws_size;

    const float* features = (const float*)d_in[0];
    const float* coords   = (const float*)d_in[1];
    const float* ce_w1 = (const float*)d_in[2];
    const float* ce_b1 = (const float*)d_in[3];
    const float* ce_w2 = (const float*)d_in[4];
    const float* ce_b2 = (const float*)d_in[5];
    const float* ce_w3 = (const float*)d_in[6];
    const float* ce_b3 = (const float*)d_in[7];
    const float* sf_w1 = (const float*)d_in[8];
    const float* sf_b1 = (const float*)d_in[9];
    const float* sf_w2 = (const float*)d_in[10];
    const float* sf_b2 = (const float*)d_in[11];
    const float* sf_w3 = (const float*)d_in[12];
    const float* sf_b3 = (const float*)d_in[13];
    const float* sf_w4 = (const float*)d_in[14];
    const float* sf_b4 = (const float*)d_in[15];

    char* ws = (char*)d_ws;
    float*    zc    = (float*)(ws);                          //    65,536 B
    _Float16* zpart = (_Float16*)(ws + 65536);               //   131,072 B
    _Float16* cpart = (_Float16*)(ws + 196608);              // 2,097,152 B
    _Float16* w2p   = (_Float16*)(ws + 2293760);             //   131,072 B
    _Float16* w3p   = (_Float16*)(ws + 2424832);             //    65,536 B
    _Float16* w4p   = (_Float16*)(ws + 2490368);             //     4,096 B

    encoder_kernel<<<B_, 256, 0, stream>>>(features, ce_w1, ce_b1, ce_w2, ce_b2,
                                           ce_w3, ce_b3, zc);
    zpart_kernel<<<B_, 256, 0, stream>>>(zc, sf_w1, sf_b1, zpart);
    cpart_kernel<<<(GG_ * HD_) / 256, 256, 0, stream>>>(coords, sf_w1, cpart);
    pack_b_kernel<<<(256 * 256) / 256, 256, 0, stream>>>(sf_w2, w2p, 256, 256);
    pack_b_kernel<<<(256 * 128) / 256, 256, 0, stream>>>(sf_w3, w3p, 256, 128);
    pack_w4_kernel<<<8, 256, 0, stream>>>(sf_w4, w4p);

    field_kernel<<<NWG_, 256, SM_TOTAL, stream>>>(cpart, zpart, w2p, w3p, w4p,
                                                  sf_b2, sf_b3, sf_b4,
                                                  (float*)d_out);
}
